// LSTM_Seq2Seq_3874060501292
// MI455X (gfx1250) — matrix operand order repaired and checked by reading, not compiled, not run on hardware
//
#include <hip/hip_runtime.h>
#include <hip/hip_bf16.h>
#include <hip/hip_fp16.h>
#include <math.h>

typedef __attribute__((ext_vector_type(16))) _Float16 v16h;
typedef __attribute__((ext_vector_type(8)))  float    v8f;

#define B_     256
#define H_     256
#define TCTX_  512
#define TPRED_ 128
#define MT_    16      // batch rows per workgroup
#define AS0W_  296     // As0 row: [input(8) | h0(256) | zero tail(32)]  (16B-aligned, 37-dword stride)
#define AS1W_  264     // As1 row: [h1(256) | pad(8)]                    (16B-aligned, 33-dword stride)

// workspace layout: packed f16 weights then f32 biases
// layer0 packed: KT=9  ktiles * 64 ntiles * 512 halves = 294912 halves
// layer1 packed: KT=16 ktiles * 64 ntiles * 512 halves = 524288 halves
#define M0H 294912
#define M1H 524288
#define OFF_ENC_W0_H 0
#define OFF_ENC_W1_H (M0H)
#define OFF_DEC_W0_H (M0H + M1H)
#define OFF_DEC_W1_H (2*M0H + M1H)
#define TOT_W_H      (2*(M0H + M1H))        // 1,638,400 halves = 3,276,800 bytes
#define OFF_BIAS_B   (2u*(M0H + M1H)*2u)    // byte offset of 4*1024 f32 biases

// ---------------------------------------------------------------------------
// Prep: pack [Wih | Whh] (f32) into WMMA B-fragment f16 layout.
// Fragment (ktile, ntile): 32 lanes * 16 halves; lane l -> n = ntile*16 + (l&15),
// half e -> k = ktile*32 + (l>>4)*16 + e.  B[k][n] = Wcomb[n][k].
// ---------------------------------------------------------------------------
__global__ void pack_weights_kernel(const float* eWih0, const float* eWhh0,
                                    const float* eWih1, const float* eWhh1,
                                    const float* dWih0, const float* dWhh0,
                                    const float* dWih1, const float* dWhh1,
                                    _Float16* outp)
{
    int idx = blockIdx.x * blockDim.x + threadIdx.x;
    if (idx >= TOT_W_H) return;
    const float *Wih, *Whh; int layer, rel, base;
    if (idx < M0H)            { Wih=eWih0; Whh=eWhh0; layer=0; rel=idx;             base=OFF_ENC_W0_H; }
    else if (idx < M0H+M1H)   { Wih=eWih1; Whh=eWhh1; layer=1; rel=idx-M0H;         base=OFF_ENC_W1_H; }
    else if (idx < 2*M0H+M1H) { Wih=dWih0; Whh=dWhh0; layer=0; rel=idx-M0H-M1H;     base=OFF_DEC_W0_H; }
    else                      { Wih=dWih1; Whh=dWhh1; layer=1; rel=idx-2*M0H-M1H;   base=OFF_DEC_W1_H; }
    int e     = rel & 15;
    int lane  = (rel >> 4) & 31;
    int ntile = (rel >> 9) & 63;
    int kt    = rel >> 15;
    int k = kt*32 + ((lane >> 4) << 4) + e;
    int n = (ntile << 4) + (lane & 15);
    float v;
    if (layer == 0) v = (k < 8) ? Wih[n*8 + k] : ((k < 264) ? Whh[n*256 + (k-8)] : 0.0f);
    else            v = (k < 256) ? Wih[n*256 + k] : Whh[n*256 + (k-256)];
    outp[base + rel] = (_Float16)v;
}

__global__ void prep_bias_kernel(const float* eb0a, const float* eb0b,
                                 const float* eb1a, const float* eb1b,
                                 const float* db0a, const float* db0b,
                                 const float* db1a, const float* db1b,
                                 float* outb)
{
    int i = blockIdx.x * blockDim.x + threadIdx.x;
    if (i >= 4096) return;
    int s = i >> 10, n = i & 1023;
    const float *a, *b;
    switch (s) {
        case 0:  a = eb0a; b = eb0b; break;
        case 1:  a = eb1a; b = eb1b; break;
        case 2:  a = db0a; b = db0b; break;
        default: a = db1a; b = db1b; break;
    }
    outb[i] = a[n] + b[n];
}

// ---------------------------------------------------------------------------
// Main recurrent kernel
// ---------------------------------------------------------------------------
// Fast branch-free activations on the v_exp_f32 / v_rcp_f32 path (~1 ulp,
// far below the f16 WMMA noise floor).  rcp(inf)=0, rcp(1)=1 keep the
// asymptotes exact; no NaNs for finite gates.
__device__ __forceinline__ float sigm_(float x) {
    return __builtin_amdgcn_rcpf(1.0f + __expf(-x));
}
__device__ __forceinline__ float tanh_(float x) {
    return 1.0f - 2.0f * __builtin_amdgcn_rcpf(__expf(2.0f*x) + 1.0f);
}

// Accumulate KT k-tiles: acc[t] += A(Arow) x B(wp).  Arow = this lane's A-matrix
// row base in LDS (unconditional, aligned 16B groups).  All 8 N-tile B
// fragments are loaded as one clause, then 8 WMMAs run back-to-back.
template<int KT>
__device__ __forceinline__ void mm_accum(v8f* acc, const _Float16* Arow,
                                         const uint4* __restrict__ wp,
                                         int w, int l)
{
    const int lh = l >> 4;
    #pragma unroll 1
    for (int kt = 0; kt < KT; ++kt) {
        union { v16h h; uint4 q[2]; } A;
        A.q[0] = *(const uint4*)(Arow + kt*32 + 16*lh);
        A.q[1] = *(const uint4*)(Arow + kt*32 + 16*lh + 8);
        union { v16h h; uint4 q[2]; } Bf[8];
        const uint4* pk = wp + ((size_t)kt * 2048 + l) * 2;   // kt * 64ntiles * 32lanes
        #pragma unroll
        for (int t = 0; t < 8; ++t) {
            int ntile = ((t >> 1) << 4) + (w << 1) + (t & 1);
            const uint4* p = pk + ntile * 64;                 // 32 lanes * 2 uint4
            Bf[t].q[0] = p[0];
            Bf[t].q[1] = p[1];
        }
        #pragma unroll
        for (int t = 0; t < 8; ++t)
            acc[t] = __builtin_amdgcn_wmma_f32_16x16x32_f16(
                false, A.h, false, Bf[t].h, (short)0, acc[t], false, false);
    }
}

// Register-local LSTM cell: lane l holds gates for n = w*32 + j16*16 + (l&15),
// rows m = 8*(l>>4)+r.  Writes h_new (f16) into hout[m*rowStride + colOff + j].
__device__ __forceinline__ void lstm_cell(v8f* acc, float* creg,
                                          _Float16* hout, int rowStride, int colOff,
                                          int w, int l)
{
    const int m  = l & 15;
    const int lh = l >> 4;
    #pragma unroll
    for (int j16 = 0; j16 < 2; ++j16) {
        #pragma unroll
        for (int r = 0; r < 8; ++r) {
            float gi = acc[0 + j16][r];
            float gf = acc[2 + j16][r];
            float gg = acc[4 + j16][r];
            float go = acc[6 + j16][r];
            float c  = creg[j16*8 + r];
            float cn = sigm_(gf) * c + sigm_(gi) * tanh_(gg);
            creg[j16*8 + r] = cn;
            float hn = sigm_(go) * tanh_(cn);
            int mm = (lh << 3) + r;
            int j  = (w << 5) + (j16 << 4) + m;
            hout[mm * rowStride + colOff + j] = (_Float16)hn;
        }
    }
}

__global__ void __launch_bounds__(256, 1)
lstm_seq2seq_kernel(const float* __restrict__ xc, const float* __restrict__ yc,
                    const float* __restrict__ xt, const float* __restrict__ yt,
                    const unsigned char* __restrict__ ws,
                    const float* __restrict__ fcw, const float* __restrict__ fcb,
                    float* __restrict__ out)
{
    __shared__ __align__(16) _Float16 As0[MT_][AS0W_];  // [in(8) | h0(256) | 0s]
    __shared__ __align__(16) _Float16 As1[MT_][AS1W_];  // [h1(256) | pad]

    const int tid   = threadIdx.x;
    const int w     = tid >> 5;
    const int l     = tid & 31;
    const int m     = l & 15;
    const int bbase = blockIdx.x * MT_;

    for (int i = tid; i < MT_*AS0W_; i += 256) ((_Float16*)As0)[i] = (_Float16)0.0f;
    for (int i = tid; i < MT_*AS1W_; i += 256) ((_Float16*)As1)[i] = (_Float16)0.0f;
    float c0[16], c1[16];
    #pragma unroll
    for (int i = 0; i < 16; ++i) { c0[i] = 0.0f; c1[i] = 0.0f; }
    __syncthreads();

    const float* bias = (const float*)(ws + OFF_BIAS_B);
    const float  fcb0 = fcb[0];
    const _Float16* A0row = &As0[m][0];
    const _Float16* A1row = &As1[m][0];

    #pragma unroll 1
    for (int phase = 0; phase < 2; ++phase) {
        const uint4* W0 = (const uint4*)((const _Float16*)ws +
                              (phase ? OFF_DEC_W0_H : OFF_ENC_W0_H));
        const uint4* W1 = (const uint4*)((const _Float16*)ws +
                              (phase ? OFF_DEC_W1_H : OFF_ENC_W1_H));
        const uint4* W1b = W1 + 32768;       // k-tiles 8..15 (8*64*32 frags * 2 uint4)
        const float* b0 = bias + (phase ? 2048 : 0);
        const float* b1 = b0 + 1024;
        const int TT = phase ? TPRED_ : TCTX_;

        float bias0r[8], bias1r[8];
        #pragma unroll
        for (int t = 0; t < 8; ++t) {
            int q = t >> 1, j16 = t & 1;
            int n = q*256 + w*32 + j16*16 + m;
            bias0r[t] = b0[n];
            bias1r[t] = b1[n];
        }

        #pragma unroll 1
        for (int t = 0; t < TT; ++t) {
            if (tid < 128) {                       // stage merged input row, f16
                int mm = tid >> 3, f = tid & 7;
                size_t b = (size_t)(bbase + mm);
                float v;
                if (!phase)       v = (f < 7) ? xc[(b*TCTX_ + t)*7 + f]       : yc[b*TCTX_ + t];
                else if (t == 0)  v = (f < 7) ? xc[(b*TCTX_ + TCTX_-1)*7 + f] : 0.0f;
                else              v = (f < 7) ? xt[(b*TPRED_ + t-1)*7 + f]    : yt[b*TPRED_ + t-1];
                As0[mm][f] = (_Float16)v;
            }
            __syncthreads();                       // input visible

            // ---- layer 0: gates = [in|h0_prev] x W0  (K=288, 9 k-tiles)
            v8f acc[8];
            #pragma unroll
            for (int q = 0; q < 8; ++q) {
                float bv = bias0r[q];
                #pragma unroll
                for (int r = 0; r < 8; ++r) acc[q][r] = bv;
            }
            mm_accum<9>(acc, A0row, W0, w, l);
            __syncthreads();                       // all reads of h0_prev done
            lstm_cell(acc, c0, &As0[0][0], AS0W_, 8, w, l);
            __syncthreads();                       // h0_new visible

            // ---- layer 1: gates = [h0_new|h1_prev] x W1  (K=512, 8+8 k-tiles)
            #pragma unroll
            for (int q = 0; q < 8; ++q) {
                float bv = bias1r[q];
                #pragma unroll
                for (int r = 0; r < 8; ++r) acc[q][r] = bv;
            }
            mm_accum<8>(acc, A0row + 8, W1,  w, l);   // h0_new part
            mm_accum<8>(acc, A1row,     W1b, w, l);   // h1_prev part
            __syncthreads();                       // all reads of h1_prev done
            lstm_cell(acc, c1, &As1[0][0], AS1W_, 0, w, l);
            __syncthreads();                       // h1_new visible

            if (phase && tid < 16) {               // fc head: DY=1 dot per row
                float s = fcb0;
                for (int j = 0; j < H_; ++j) s += (float)As1[tid][j] * fcw[j];
                out[(size_t)(bbase + tid) * TPRED_ + t] = s;
            }
        }
    }
}

// ---------------------------------------------------------------------------
extern "C" void kernel_launch(void* const* d_in, const int* in_sizes, int n_in,
                              void* d_out, int out_size, void* d_ws, size_t ws_size,
                              hipStream_t stream)
{
    const float* xc = (const float*)d_in[0];
    const float* yc = (const float*)d_in[1];
    const float* xt = (const float*)d_in[2];
    const float* yt = (const float*)d_in[3];
    // d_in[4] = teacher_forcing_prob (==1, always teacher-forced)
    const float* eWih0 = (const float*)d_in[5];
    const float* eWhh0 = (const float*)d_in[6];
    const float* ebih0 = (const float*)d_in[7];
    const float* ebhh0 = (const float*)d_in[8];
    const float* eWih1 = (const float*)d_in[9];
    const float* eWhh1 = (const float*)d_in[10];
    const float* ebih1 = (const float*)d_in[11];
    const float* ebhh1 = (const float*)d_in[12];
    const float* dWih0 = (const float*)d_in[13];
    const float* dWhh0 = (const float*)d_in[14];
    const float* dbih0 = (const float*)d_in[15];
    const float* dbhh0 = (const float*)d_in[16];
    const float* dWih1 = (const float*)d_in[17];
    const float* dWhh1 = (const float*)d_in[18];
    const float* dbih1 = (const float*)d_in[19];
    const float* dbhh1 = (const float*)d_in[20];
    const float* fcW   = (const float*)d_in[21];
    const float* fcb   = (const float*)d_in[22];

    unsigned char* ws = (unsigned char*)d_ws;

    pack_weights_kernel<<<(TOT_W_H + 255) / 256, 256, 0, stream>>>(
        eWih0, eWhh0, eWih1, eWhh1, dWih0, dWhh0, dWih1, dWhh1, (_Float16*)ws);
    prep_bias_kernel<<<16, 256, 0, stream>>>(
        ebih0, ebhh0, ebih1, ebhh1, dbih0, dbhh0, dbih1, dbhh1,
        (float*)(ws + OFF_BIAS_B));
    lstm_seq2seq_kernel<<<B_ / MT_, 256, 0, stream>>>(
        xc, yc, xt, yt, ws, fcW, fcb, (float*)d_out);
}
